// MSAColumnAttention_79345225826826
// MI455X (gfx1250) — compile-verified, matrix operand order checked by reading, not run on hardware
//
#include <hip/hip_runtime.h>
#include <hip/hip_bf16.h>

// ---------------------------------------------------------------------------
// Problem: B=2, L=512, D=256, P=128
//   r = msa @ W_row + b_row                          [B,L,D]
//   out[b,i,j,p] = (sum_d r[b,i,d] r[b,j,d] W_out[d,p] + b_out[p]) * m_i m_j
// Factorization: H[(b,i,j),d] = r[b,i,d]*r[b,j,d];  out = H @ W_out  (+bias,mask)
// bf16 hi/lo split GEMM on v_wmma_f32_16x16x32_bf16 -> ~fp32 accuracy,
// memory-bound on the 268MB output write (~11.5us floor at 23.3 TB/s).
// This revision: dual N-tile / dual-accumulator inner loop so the matrix pipe
// sees two independent WMMA chains instead of one serialized RAW chain.
// ---------------------------------------------------------------------------

typedef __attribute__((ext_vector_type(16))) __bf16 v16bf;
typedef __attribute__((ext_vector_type(8)))  float  v8f;
typedef __attribute__((ext_vector_type(4)))  float  f32x4;

#define Lseq 512
#define Ddim 256
#define Pdim 128
#define WSTRIDE 264   // bf16 elems per W row in LDS; 132 dwords/row, %64=4 -> conflict-free b128
#define RJSTRIDE 260  // floats per rj row in LDS; %64=4 -> conflict-free b128
// LDS layout (bytes):
//   Whi : 128*264*2 = 67584
//   Wlo : 67584                     (offset 67584)
//   rjL : 128*260*4 = 133120        (offset 135168)
//   riL : 256*4 = 1024              (offset 268288)
#define LDS_BYTES (135168 + 133120 + 1024)

// ---------------- Kernel 1: row projection r = msa @ W_row + b_row ----------
__global__ __launch_bounds__(256)
void msa_rowproj_kernel(const float* __restrict__ msa,
                        const float* __restrict__ Wrow,
                        const float* __restrict__ brow,
                        float* __restrict__ r) {
    __shared__ float rowbuf[Ddim];
    const int bl = blockIdx.x;           // b*L + l  (0..1023)
    const int t  = threadIdx.x;          // output d (0..255)
    rowbuf[t] = msa[(size_t)bl * Ddim + t];
    __syncthreads();
    float acc = brow[t];
#pragma unroll 4
    for (int k = 0; k < Ddim; ++k)
        acc += rowbuf[k] * Wrow[(size_t)k * Ddim + t];   // coalesced across t
    r[(size_t)bl * Ddim + t] = acc;
}

// ---------------- Kernel 2: pair outer-product GEMM with WMMA ---------------
__global__ __launch_bounds__(256)
void pair_outer_wmma_kernel(const float* __restrict__ r,
                            const unsigned char* __restrict__ mask,
                            const float* __restrict__ Wout,
                            const float* __restrict__ bout,
                            float* __restrict__ out) {
    extern __shared__ char lds[];
    __bf16* Whi = (__bf16*)lds;                       // [128][WSTRIDE] transposed: [p][d]
    __bf16* Wlo = Whi + 128 * WSTRIDE;
    float*  rjL = (float*)(lds + 135168);             // [128][RJSTRIDE]
    float*  riL = rjL + 128 * RJSTRIDE;               // [256]

    const int tid   = threadIdx.x;
    const int jblk  = blockIdx.x;        // 0..3  (128 j's each)
    const int i     = blockIdx.y;        // 0..511
    const int b     = blockIdx.z;        // 0..1
    const int jbase0 = jblk * 128;

    const int lane  = tid & 31;
    const int wave  = tid >> 5;          // 8 waves; wave owns 16 j-rows x 128 p
    const int mrow  = lane & 15;         // M row / N col within a 16x16 tile
    const int khalf = lane >> 4;         // K-half select per WMMA 16-bit layout

    const size_t outBase = (((size_t)(b * Lseq + i)) * Lseq) * Pdim;
    const int jbaseW = jbase0 + wave * 16;

    const unsigned char mi = mask[b * Lseq + i];   // block-uniform
    if (!mi) {
        // whole 128x128 tile is zero: fast store path (no WMMA, uniform branch)
        for (int g = 0; g < 16; ++g) {
            size_t rowoff = outBase + (size_t)(jbaseW + g) * Pdim;
            for (int p = lane; p < Pdim; p += 32) out[rowoff + p] = 0.0f;
        }
        return;
    }

    // ---- stage W_out (split+transposed) and the 128 r_j rows into LDS ----
    for (int idx = tid; idx < Ddim * Pdim; idx += 256) {
        int d = idx >> 7, p = idx & 127;
        float w = Wout[idx];                          // coalesced [d][p]
        __bf16 wh = (__bf16)w;
        Whi[p * WSTRIDE + d] = wh;
        Wlo[p * WSTRIDE + d] = (__bf16)(w - (float)wh);
    }
    for (int idx = tid; idx < 128 * Ddim; idx += 256) {
        int jl = idx >> 8, k = idx & 255;
        rjL[jl * RJSTRIDE + k] = r[((size_t)(b * Lseq) + jbase0 + jl) * Ddim + k];
    }
    riL[tid] = r[((size_t)(b * Lseq) + i) * Ddim + tid];
    __syncthreads();

    // ---- build all 8 A fragments: H = r_i .* r_j, split into bf16 hi/lo ----
    // 16-bit A 16x32 layout: lane&15 = M row; lane>>4 picks K 0..7/16..23 vs 8..15/24..31
    const float* rjrow = &rjL[(wave * 16 + mrow) * RJSTRIDE];
    v16bf Ahi[8], Alo[8];
#pragma unroll
    for (int s = 0; s < 8; ++s) {
        const int k1 = s * 32 + khalf * 8;
        f32x4 ri0 = *(const f32x4*)&riL[k1];
        f32x4 ri1 = *(const f32x4*)&riL[k1 + 4];
        f32x4 ri2 = *(const f32x4*)&riL[k1 + 16];
        f32x4 ri3 = *(const f32x4*)&riL[k1 + 20];
        f32x4 rj0 = *(const f32x4*)&rjrow[k1];
        f32x4 rj1 = *(const f32x4*)&rjrow[k1 + 4];
        f32x4 rj2 = *(const f32x4*)&rjrow[k1 + 16];
        f32x4 rj3 = *(const f32x4*)&rjrow[k1 + 20];
#pragma unroll
        for (int e = 0; e < 4; ++e) {
            float h; __bf16 x;
            h = ri0[e] * rj0[e]; x = (__bf16)h; Ahi[s][e]      = x; Alo[s][e]      = (__bf16)(h - (float)x);
            h = ri1[e] * rj1[e]; x = (__bf16)h; Ahi[s][e + 4]  = x; Alo[s][e + 4]  = (__bf16)(h - (float)x);
            h = ri2[e] * rj2[e]; x = (__bf16)h; Ahi[s][e + 8]  = x; Alo[s][e + 8]  = (__bf16)(h - (float)x);
            h = ri3[e] * rj3[e]; x = (__bf16)h; Ahi[s][e + 12] = x; Alo[s][e + 12] = (__bf16)(h - (float)x);
        }
    }

    // ---- main loop: 2 N-tiles per iteration, two independent WMMA chains ---
    union BFrag { v16bf v; f32x4 f[2]; };
#pragma unroll 1
    for (int nt = 0; nt < 8; nt += 2) {
        const int p0 = nt * 16 + mrow;                // N column, tile 0
        const int p1 = p0 + 16;                       // N column, tile 1
        const __bf16* wh0 = &Whi[p0 * WSTRIDE];
        const __bf16* wl0 = &Wlo[p0 * WSTRIDE];
        const __bf16* wh1 = &Whi[p1 * WSTRIDE];
        const __bf16* wl1 = &Wlo[p1 * WSTRIDE];
        v8f acc0 = {0.f, 0.f, 0.f, 0.f, 0.f, 0.f, 0.f, 0.f};
        v8f acc1 = {0.f, 0.f, 0.f, 0.f, 0.f, 0.f, 0.f, 0.f};
#pragma unroll
        for (int s = 0; s < 8; ++s) {
            const int k1 = s * 32 + khalf * 8;        // 16B aligned in LDS
            BFrag bh0, bl0, bh1, bl1;
            bh0.f[0] = *(const f32x4*)&wh0[k1];
            bh0.f[1] = *(const f32x4*)&wh0[k1 + 16];
            bh1.f[0] = *(const f32x4*)&wh1[k1];
            bh1.f[1] = *(const f32x4*)&wh1[k1 + 16];
            bl0.f[0] = *(const f32x4*)&wl0[k1];
            bl0.f[1] = *(const f32x4*)&wl0[k1 + 16];
            bl1.f[0] = *(const f32x4*)&wl1[k1];
            bl1.f[1] = *(const f32x4*)&wl1[k1 + 16];
            // interleave the two independent accumulator chains
            acc0 = __builtin_amdgcn_wmma_f32_16x16x32_bf16(false, Ahi[s], false, bh0.v,
                                                           (short)0, acc0, false, false);
            acc1 = __builtin_amdgcn_wmma_f32_16x16x32_bf16(false, Ahi[s], false, bh1.v,
                                                           (short)0, acc1, false, false);
            acc0 = __builtin_amdgcn_wmma_f32_16x16x32_bf16(false, Ahi[s], false, bl0.v,
                                                           (short)0, acc0, false, false);
            acc1 = __builtin_amdgcn_wmma_f32_16x16x32_bf16(false, Ahi[s], false, bl1.v,
                                                           (short)0, acc1, false, false);
            acc0 = __builtin_amdgcn_wmma_f32_16x16x32_bf16(false, Alo[s], false, bh0.v,
                                                           (short)0, acc0, false, false);
            acc1 = __builtin_amdgcn_wmma_f32_16x16x32_bf16(false, Alo[s], false, bh1.v,
                                                           (short)0, acc1, false, false);
        }
        // epilogue: D layout -> VGPR g holds row M = g + 8*khalf, col N = lane&15
        const float bias0 = bout[p0];
        const float bias1 = bout[p1];
#pragma unroll
        for (int g = 0; g < 8; ++g) {
            const int jrow = jbaseW + g + khalf * 8;
            const float mj = mask[b * Lseq + jrow] ? 1.0f : 0.0f;
            const size_t rowoff = outBase + (size_t)jrow * Pdim;
            out[rowoff + p0] = fmaf(acc0[g], mj, bias0 * mj);
            out[rowoff + p1] = fmaf(acc1[g], mj, bias1 * mj);
        }
    }
}

// ---------------------------------------------------------------------------
extern "C" void kernel_launch(void* const* d_in, const int* in_sizes, int n_in,
                              void* d_out, int out_size, void* d_ws, size_t ws_size,
                              hipStream_t stream) {
    // setup_inputs order: msa_repr, pair_repr(unused), mask, W_row, b_row, W_out, b_out
    const float*         msa  = (const float*)d_in[0];
    const unsigned char* mask = (const unsigned char*)d_in[2];  // jnp bool = 1 byte
    const float*         Wrow = (const float*)d_in[3];
    const float*         brow = (const float*)d_in[4];
    const float*         Wout = (const float*)d_in[5];
    const float*         bout = (const float*)d_in[6];
    float* out = (float*)d_out;
    float* r   = (float*)d_ws;   // B*L*D floats = 1 MB scratch

    // r = msa @ W_row + b_row
    msa_rowproj_kernel<<<dim3(2 * Lseq), dim3(256), 0, stream>>>(msa, Wrow, brow, r);

    // out[b,i,j,p]
    dim3 grid(4, Lseq, 2);   // j-blocks of 128, i, b
    pair_outer_wmma_kernel<<<grid, dim3(256), LDS_BYTES, stream>>>(r, mask, Wout, bout, out);
}
